// DynamicGraphLearning_24919400251997
// MI455X (gfx1250) — matrix operand order repaired and checked by reading, not compiled, not run on hardware
//
#include <hip/hip_runtime.h>
#include <hip/hip_bf16.h>
#include <cstdint>

typedef __attribute__((ext_vector_type(16))) _Float16 v16h;
typedef __attribute__((ext_vector_type(8)))  float    v8f;

#define NN 50000
#define EE 800000
#define CC 128

union AFrag { v16h v; float4 f4[2]; };

__device__ __forceinline__ float sigmoidf(float z) { return 1.0f / (1.0f + __expf(-z)); }

// ---------------- utility kernels ----------------
__global__ void k_zero(float* __restrict__ p, int n) {
    int i = blockIdx.x * blockDim.x + threadIdx.x;
    if (i < n) p[i] = 0.0f;
}

__global__ void k_cvt_f16(const float* __restrict__ src, _Float16* __restrict__ dst, int n) {
    int i = blockIdx.x * blockDim.x + threadIdx.x;
    if (i < n) dst[i] = (_Float16)src[i];
}

__global__ void k_dinv(const float* __restrict__ deg, float* __restrict__ dinv, int n) {
    int i = blockIdx.x * blockDim.x + threadIdx.x;
    if (i < n) {
        float d = deg[i];
        dinv[i] = (d > 0.0f) ? rsqrtf(d) : 0.0f;
    }
}

__global__ void k_norm(const int* __restrict__ eidx, const float* __restrict__ ew,
                       const float* __restrict__ dinv, float* __restrict__ nrm) {
    int e = blockIdx.x * blockDim.x + threadIdx.x;
    if (e < EE) {
        int r = eidx[e];
        int c = eidx[EE + e];
        nrm[e] = -dinv[r] * ew[e] * dinv[c];
    }
}

// scatter SpMM: outacc[col[e]] += nrm[e] * vin[row[e]]   (4 channels per thread)
__global__ void k_spmm(const float* __restrict__ vin, const int* __restrict__ eidx,
                       const float* __restrict__ nrm, float* __restrict__ outacc) {
    int t = blockIdx.x * blockDim.x + threadIdx.x;
    int e = t >> 5;
    if (e >= EE) return;
    int c4 = (t & 31) * 4;
    float w = nrm[e];
    int r = eidx[e];
    int c = eidx[EE + e];
    const float4 xv = *(const float4*)(vin + (size_t)r * CC + c4);
    float* dst = outacc + (size_t)c * CC + c4;
    atomicAdd(dst + 0, w * xv.x);
    atomicAdd(dst + 1, w * xv.y);
    atomicAdd(dst + 2, w * xv.z);
    atomicAdd(dst + 3, w * xv.w);
}

// tx2 = 2*tx2 - x
__global__ void k_fin_tx2(float* __restrict__ tx2, const float* __restrict__ x, int n) {
    int i = blockIdx.x * blockDim.x + threadIdx.x;
    if (i < n) tx2[i] = 2.0f * tx2[i] - x[i];
}

// ---------------- edge MLP (WMMA) ----------------
// Per wave: TWO 16-edge tiles sharing each B fragment (halves LDS bytes/WMMA so
// the XDL pipe, not LDS bandwidth, is the limiter).
// A = gathered [16 x 256] f16 (x[row] ++ x[col]), B = w1 [256 x 128] f16 in LDS.
// Fused epilogue: +b1, ReLU, dot w2, +b2, sigmoid, store ew, atomic deg.
__launch_bounds__(256, 1)
__global__ void k_edge_mlp(const _Float16* __restrict__ xh,
                           const int* __restrict__ eidx,
                           const float* __restrict__ w1,   // [256][128] f32 row-major
                           const float* __restrict__ b1,   // [128]
                           const float* __restrict__ w2,   // [128]
                           const float* __restrict__ b2p,  // [1]
                           float* __restrict__ ew,         // [E]
                           float* __restrict__ deg)        // [N]
{
    __shared__ _Float16 w1T[128 * 264];  // [n][k] padded stride (33*8) -> conflict-free 16B reads
    for (int i = threadIdx.x; i < 256 * 128; i += 256) {
        int k = i >> 7, n = i & 127;
        w1T[n * 264 + k] = (_Float16)w1[i];
    }
    __syncthreads();

    const int lane = threadIdx.x & 31;
    const int wave = threadIdx.x >> 5;
    const int pair = blockIdx.x * 8 + wave;  // processes edges [pair*32, pair*32+32)
    const int e_base = pair * 32;
    if (e_base >= EE) return;

    const int m  = lane & 15;   // edge within tile (A row)
    const int hi = lane >> 4;   // half-wave select

    int e0 = e_base + m;
    int e1 = e_base + 16 + m;
    if (e0 >= EE) e0 = EE - 1;
    if (e1 >= EE) e1 = EE - 1;
    const int r0 = eidx[e0];
    const int c0 = eidx[EE + e0];
    const int r1 = eidx[e1];
    const int c1 = eidx[EE + e1];
    const _Float16* arow0 = xh + (size_t)r0 * CC;
    const _Float16* acol0 = xh + (size_t)c0 * CC;
    const _Float16* arow1 = xh + (size_t)r1 * CC;
    const _Float16* acol1 = xh + (size_t)c1 * CC;

    float b1v[8], w2v[8];
#pragma unroll
    for (int nt = 0; nt < 8; ++nt) {
        int n = nt * 16 + m;
        b1v[nt] = b1[n];
        w2v[nt] = w2[n];
    }
    const float b2 = b2p[0];

    v8f acc0[8] = {};
    v8f acc1[8] = {};

#pragma unroll
    for (int kb = 0; kb < 8; ++kb) {            // K = 256 in steps of 32
        const _Float16* src0 = (kb < 4) ? arow0 : acol0;
        const _Float16* src1 = (kb < 4) ? arow1 : acol1;
        const int koff = (kb & 3) * 32 + hi * 8;
        AFrag a0, a1;
        a0.f4[0] = *(const float4*)(src0 + koff);        // K = base + hi*8 .. +7
        a0.f4[1] = *(const float4*)(src0 + koff + 16);   // K = base + 16 + hi*8 .. +7
        a1.f4[0] = *(const float4*)(src1 + koff);
        a1.f4[1] = *(const float4*)(src1 + koff + 16);
#pragma unroll
        for (int nt = 0; nt < 8; ++nt) {        // N = 128 in tiles of 16
            const int n = nt * 16 + m;
            AFrag b;
            const _Float16* bp = &w1T[n * 264 + kb * 32 + hi * 8];
            b.f4[0] = *(const float4*)(bp);
            b.f4[1] = *(const float4*)(bp + 16);
            acc0[nt] = __builtin_amdgcn_wmma_f32_16x16x32_f16(
                false, a0.v, false, b.v, (short)0, acc0[nt], false, false);
            acc1[nt] = __builtin_amdgcn_wmma_f32_16x16x32_f16(
                false, a1.v, false, b.v, (short)0, acc1[nt], false, false);
        }
    }

    // epilogue: relu(acc + b1) . w2  -> per-edge scalar, for both tiles
    float s0[8], s1[8];
#pragma unroll
    for (int v = 0; v < 8; ++v) { s0[v] = 0.0f; s1[v] = 0.0f; }
#pragma unroll
    for (int nt = 0; nt < 8; ++nt) {
#pragma unroll
        for (int v = 0; v < 8; ++v) {
            float h0 = fmaxf(acc0[nt][v] + b1v[nt], 0.0f);
            float h1 = fmaxf(acc1[nt][v] + b1v[nt], 0.0f);
            s0[v] += h0 * w2v[nt];
            s1[v] += h1 * w2v[nt];
        }
    }
    // reduce over the 16 lanes of each half-wave (n dimension)
#pragma unroll
    for (int v = 0; v < 8; ++v) {
        s0[v] += __shfl_xor(s0[v], 1, 32);
        s0[v] += __shfl_xor(s0[v], 2, 32);
        s0[v] += __shfl_xor(s0[v], 4, 32);
        s0[v] += __shfl_xor(s0[v], 8, 32);
        s1[v] += __shfl_xor(s1[v], 1, 32);
        s1[v] += __shfl_xor(s1[v], 2, 32);
        s1[v] += __shfl_xor(s1[v], 4, 32);
        s1[v] += __shfl_xor(s1[v], 8, 32);
    }
    // lane 0 holds edges base+0..7 ; lane 16 holds base+8..15 (per tile)
#pragma unroll
    for (int v = 0; v < 8; ++v) {
        int src_lane = (hi ? 8 : 0) + v;        // lane holding that edge's row index
        float val0 = sigmoidf(s0[v] + b2);
        float val1 = sigmoidf(s1[v] + b2);
        int rv0 = __shfl(r0, src_lane, 32);
        int rv1 = __shfl(r1, src_lane, 32);
        int eo0 = e_base + hi * 8 + v;
        int eo1 = e_base + 16 + hi * 8 + v;
        if (m == 0 && eo0 < EE) {
            ew[eo0] = val0;
            atomicAdd(&deg[rv0], val0);
        }
        if (m == 0 && eo1 < EE) {
            ew[eo1] = val1;
            atomicAdd(&deg[rv1], val1);
        }
    }
}

// ---------------- final Chebyshev GEMM (WMMA) ----------------
// out = tx0@W0 + tx1@W1 + tx2@W2 + bias.  Per wave: 16 nodes x 128 out.
__launch_bounds__(256, 1)
__global__ void k_cheb_out(const _Float16* __restrict__ xh,
                           const float* __restrict__ tx1,
                           const float* __restrict__ tx2,
                           const float* __restrict__ wch,  // [3][128][128] f32
                           const float* __restrict__ bias, // [128]
                           float* __restrict__ out)        // [N][128] f32
{
    __shared__ _Float16 wT[3 * 128 * 136];  // [t][n][k], padded stride (17*8)
    for (int i = threadIdx.x; i < 3 * 128 * 128; i += 256) {
        int t = i >> 14;
        int rem = i & 16383;
        int k = rem >> 7, n = rem & 127;
        wT[(t * 128 + n) * 136 + k] = (_Float16)wch[i];
    }
    __syncthreads();

    const int lane = threadIdx.x & 31;
    const int wave = threadIdx.x >> 5;
    const int tile = blockIdx.x * 8 + wave;
    const int nTiles = NN / 16;  // 3125
    if (tile >= nTiles) return;

    const int m  = lane & 15;
    const int hi = lane >> 4;
    const int node = tile * 16 + m;

    const _Float16* a0 = xh  + (size_t)node * CC;
    const float*    a1 = tx1 + (size_t)node * CC;
    const float*    a2 = tx2 + (size_t)node * CC;

    v8f acc[8] = {};

#pragma unroll
    for (int t = 0; t < 3; ++t) {
#pragma unroll
        for (int kb = 0; kb < 4; ++kb) {        // K = 128 per term
            const int k0 = kb * 32 + hi * 8;
            AFrag a;
            if (t == 0) {
                a.f4[0] = *(const float4*)(a0 + k0);
                a.f4[1] = *(const float4*)(a0 + k0 + 16);
            } else {
                const float* af = (t == 1) ? a1 : a2;
#pragma unroll
                for (int j = 0; j < 8; ++j) {   // f32 -> f16 in-register
                    a.v[j]     = (_Float16)af[k0 + j];
                    a.v[8 + j] = (_Float16)af[k0 + 16 + j];
                }
            }
#pragma unroll
            for (int nt = 0; nt < 8; ++nt) {
                const int n = nt * 16 + m;
                AFrag b;
                const _Float16* bp = &wT[(t * 128 + n) * 136 + kb * 32 + hi * 8];
                b.f4[0] = *(const float4*)(bp);
                b.f4[1] = *(const float4*)(bp + 16);
                acc[nt] = __builtin_amdgcn_wmma_f32_16x16x32_f16(
                    false, a.v, false, b.v, (short)0, acc[nt], false, false);
            }
        }
    }

    // epilogue: + bias, store f32. lane l, vgpr v -> node tile*16 + v + 8*hi, col nt*16+m
#pragma unroll
    for (int nt = 0; nt < 8; ++nt) {
        const int n = nt * 16 + m;
        const float bv = bias[n];
#pragma unroll
        for (int v = 0; v < 8; ++v) {
            int nodeOut = tile * 16 + v + hi * 8;
            out[(size_t)nodeOut * CC + n] = acc[nt][v] + bv;
        }
    }
}

// ---------------- launcher ----------------
extern "C" void kernel_launch(void* const* d_in, const int* in_sizes, int n_in,
                              void* d_out, int out_size, void* d_ws, size_t ws_size,
                              hipStream_t stream) {
    (void)in_sizes; (void)n_in; (void)out_size; (void)ws_size;

    const float* x    = (const float*)d_in[0];
    const int*   eidx = (const int*)d_in[1];
    const float* w1   = (const float*)d_in[2];
    const float* b1   = (const float*)d_in[3];
    const float* w2   = (const float*)d_in[4];
    const float* b2   = (const float*)d_in[5];
    const float* wch  = (const float*)d_in[6];
    const float* bias = (const float*)d_in[7];

    float* out = (float*)d_out;              // [N*C] node output
    float* ew  = out + (size_t)NN * CC;      // [E] edge weights (second tuple element)

    char* ws = (char*)d_ws;
    _Float16* xh  = (_Float16*)(ws);                  // 12,800,000 B
    float* tx1    = (float*)(ws + 12800000);          // 25,600,000 B
    float* tx2    = (float*)(ws + 38400000);          // 25,600,000 B
    float* deg    = (float*)(ws + 64000000);          //    200,000 B
    float* dinv   = (float*)(ws + 64200192);          //    200,000 B
    float* nrm    = (float*)(ws + 64400384);          //  3,200,000 B

    const int NC = NN * CC;  // 6,400,000

    k_zero<<<(NC + 255) / 256, 256, 0, stream>>>(tx1, NC);
    k_zero<<<(NC + 255) / 256, 256, 0, stream>>>(tx2, NC);
    k_zero<<<(NN + 255) / 256, 256, 0, stream>>>(deg, NN);
    k_cvt_f16<<<(NC + 255) / 256, 256, 0, stream>>>(x, xh, NC);

    // edge MLP: 800000/32 = 25000 wave-pairs, 8 per block -> 3125 blocks
    k_edge_mlp<<<EE / 256, 256, 0, stream>>>(xh, eidx, w1, b1, w2, b2, ew, deg);

    k_dinv<<<(NN + 255) / 256, 256, 0, stream>>>(deg, dinv, NN);
    k_norm<<<(EE + 255) / 256, 256, 0, stream>>>(eidx, ew, dinv, nrm);

    // tx1 = L_hat x ; tx2 = L_hat tx1 (scatter), then tx2 = 2*tx2 - x
    k_spmm<<<(EE * 32) / 256, 256, 0, stream>>>(x,   eidx, nrm, tx1);
    k_spmm<<<(EE * 32) / 256, 256, 0, stream>>>(tx1, eidx, nrm, tx2);
    k_fin_tx2<<<(NC + 255) / 256, 256, 0, stream>>>(tx2, x, NC);

    // out = tx0@W0 + tx1@W1 + tx2@W2 + bias : 3125 tiles, 8/block -> 391 blocks
    k_cheb_out<<<(3125 + 7) / 8, 256, 0, stream>>>(xh, tx1, tx2, wch, bias, out);
}
